// AttentionBlock_72954314489917
// MI455X (gfx1250) — compile-verified, matrix operand order checked by reading, not run on hardware
//
#include <hip/hip_runtime.h>
#include <hip/hip_bf16.h>

#define SEQ 4096
#define DIM 1024
#define HEADS 16
#define HD 64

typedef __bf16 bf16;
typedef __attribute__((ext_vector_type(16))) bf16 v16bf;
typedef __attribute__((ext_vector_type(8)))  bf16 v8bf;
typedef __attribute__((ext_vector_type(8)))  float v8f;

// Build a v16bf A/B operand from two 8-element (16-byte) chunks.
// Per ISA 7.12.2: elements 0..7 hold K=0..7 (lanes<16) / K=8..15 (lanes>=16);
// elements 8..15 hold K=16..23 / K=24..31.
__device__ __forceinline__ v16bf combine16(const bf16* lo, const bf16* hi) {
  v8bf a = *(const v8bf*)lo;
  v8bf b = *(const v8bf*)hi;
  v16bf r;
#pragma unroll
  for (int i = 0; i < 8; ++i) { r[i] = a[i]; r[i + 8] = b[i]; }
  return r;
}

__device__ __forceinline__ v8f zero8() {
  v8f z;
#pragma unroll
  for (int i = 0; i < 8; ++i) z[i] = 0.0f;
  return z;
}

__device__ __forceinline__ v8f wmma_bf16(v16bf a, v16bf b, v8f c) {
  return __builtin_amdgcn_wmma_f32_16x16x32_bf16(
      false, a, false, b, (short)0, c, false, false);
}

// CDNA5 async copy: global -> LDS, 16 bytes, tracked with ASYNCcnt.
__device__ __forceinline__ void issue_async_b128(unsigned int lds_off,
                                                 unsigned long long gaddr) {
  asm volatile("global_load_async_to_lds_b128 %0, %1, off"
               :: "v"(lds_off), "v"(gaddr) : "memory");
}
__device__ __forceinline__ void wait_async0() {
  asm volatile("s_wait_asynccnt 0x0" ::: "memory");
}
__device__ __forceinline__ unsigned int lds_off32(const void* p) {
  return (unsigned int)(unsigned long long)(uintptr_t)p;
}
__device__ __forceinline__ unsigned long long gaddr64(const void* p) {
  return (unsigned long long)(uintptr_t)p;
}

// ---------------------------------------------------------------------------
// Kernel 1: rotate-half RoPE on the full feature dim, downcast to bf16.
// ---------------------------------------------------------------------------
__global__ __launch_bounds__(256)
void rope_bf16_kernel(const float* __restrict__ x, const float* __restrict__ cs,
                      const float* __restrict__ sn, bf16* __restrict__ xr) {
  int idx = blockIdx.x * 256 + threadIdx.x;      // over SEQ*DIM
  int d = idx & (DIM - 1);
  float v = x[idx];
  float rot = (d < DIM / 2) ? -x[idx + DIM / 2] : x[idx - DIM / 2];
  xr[idx] = (bf16)(v * cs[idx] + rot * sn[idx]);
}

__global__ __launch_bounds__(256)
void cvt_bf16_kernel(const float* __restrict__ w, bf16* __restrict__ o) {
  int idx = blockIdx.x * 256 + threadIdx.x;
  o[idx] = (bf16)w[idx];
}

// ---------------------------------------------------------------------------
// Shared GEMM K-loop: y = x @ W^T.  Per wave one 16(M) x 64(N) tile; the B
// panel is block-uniform -> staged once per block via async-LDS copies,
// double-buffered (unroll 2 makes buffer parity static and lets the A-operand
// rotation be SSA-renamed instead of copied).  All four B operands are read
// from LDS before the WMMA burst so DS latency overlaps the matrix pipe.
// ---------------------------------------------------------------------------
__device__ __forceinline__ void gemm_loop(const bf16* __restrict__ Arow,
                                          const bf16* __restrict__ W,
                                          int nbase, bf16* sbuf, v8f acc[4]) {
  const int tid  = threadIdx.x;
  const int lane = tid & 31;
  const int hi   = lane >> 4;

  // This thread's 16B fill chunk: chunk c=tid -> tile ft, lane fl, half fh.
  const int ft = tid >> 6;
  const int fr = tid & 63;
  const int fl = fr >> 1;
  const int fh = fr & 1;
  const bf16* fsrc = W + (nbase + ft * 16 + (fl & 15)) * DIM
                       + (fl >> 4) * 8 + fh * 16;

  issue_async_b128(lds_off32(sbuf + tid * 8), gaddr64(fsrc));
  v16bf a = combine16(Arow + hi * 8, Arow + hi * 8 + 16);

#pragma unroll 2
  for (int it = 0; it < DIM / 32; ++it) {
    const int kb = it * 32;
    bf16* cur = sbuf + (it & 1) * 2048;
    wait_async0();            // own chunk landed
    __syncthreads();          // everyone's chunk landed; prev reads done
    if (kb + 32 < DIM) {
      bf16* nxt = sbuf + ((it + 1) & 1) * 2048;
      issue_async_b128(lds_off32(nxt + tid * 8), gaddr64(fsrc + kb + 32));
    }
    v16bf an = a;
    if (kb + 32 < DIM) {      // register prefetch of next A chunk
      const bf16* p = Arow + kb + 32 + hi * 8;
      an = combine16(p, p + 16);
    }
    v16bf b[4];
#pragma unroll
    for (int t = 0; t < 4; ++t) {
      const bf16* p = cur + (t * 64 + lane * 2) * 8;
      b[t] = combine16(p, p + 8);
    }
#pragma unroll
    for (int t = 0; t < 4; ++t) acc[t] = wmma_bf16(a, b[t], acc[t]);
    a = an;
  }
}

// ---------------------------------------------------------------------------
// Kernel 2: fused QKV projection.  grid = (DIM/64, SEQ/128, 3); z selects
// Q / K / V.  V is stored transposed (Vt[d][s]) for the P*V B-operand path.
// ---------------------------------------------------------------------------
__global__ __launch_bounds__(256)
void qkv_gemm_kernel(const bf16* __restrict__ X,
                     const bf16* __restrict__ Wq, const bf16* __restrict__ Wk,
                     const bf16* __restrict__ Wv,
                     const float* __restrict__ bq, const float* __restrict__ bk,
                     const float* __restrict__ bv,
                     bf16* __restrict__ Q, bf16* __restrict__ K,
                     bf16* __restrict__ Vt) {
  __shared__ bf16 sB[2 * 2048];
  const int lane = threadIdx.x & 31;
  const int wave = threadIdx.x >> 5;
  const int hi   = lane >> 4;
  const int l16  = lane & 15;
  const int z    = blockIdx.z;
  const bf16*  W    = (z == 0) ? Wq : (z == 1) ? Wk : Wv;
  const float* bias = (z == 0) ? bq : (z == 1) ? bk : bv;

  const int mtile = blockIdx.y * 128 + wave * 16;
  const int nbase = blockIdx.x * 64;

  v8f acc[4];
#pragma unroll
  for (int t = 0; t < 4; ++t) acc[t] = zero8();

  gemm_loop(X + (mtile + l16) * DIM, W, nbase, sB, acc);

#pragma unroll
  for (int t = 0; t < 4; ++t) {
    const int col = nbase + t * 16 + l16;
    const float bb = bias[col];
    if (z == 2) {
#pragma unroll
      for (int i = 0; i < 8; ++i) {
        const int row = mtile + i + hi * 8;
        Vt[col * SEQ + row] = (bf16)(acc[t][i] + bb);
      }
    } else {
      bf16* O = (z == 0) ? Q : K;
#pragma unroll
      for (int i = 0; i < 8; ++i) {
        const int row = mtile + i + hi * 8;
        O[row * DIM + col] = (bf16)(acc[t][i] + bb);
      }
    }
  }
}

// ---------------------------------------------------------------------------
// Kernel 3: flash attention with online softmax.  grid = (SEQ/128, HEADS).
// K/V tiles for each 32-key step are block-uniform -> async-LDS staged,
// double-buffered.  K and V operands are read from LDS up front so their DS
// latency hides under the f32 softmax VALU burst (XDL WMMAs co-execute).
// ---------------------------------------------------------------------------
__global__ __launch_bounds__(256)
void flash_attn_kernel(const bf16* __restrict__ Q, const bf16* __restrict__ K,
                       const bf16* __restrict__ Vt, const int* __restrict__ msk,
                       bf16* __restrict__ Aout) {
  __shared__ bf16 skv[2][512 * 8];        // 2 x 8KB: chunks 0..255 K, 256..511 V
  __shared__ bf16 plds[8][16 * 32];       // per-wave P staging
  const int tid   = threadIdx.x;
  const int lane  = tid & 31;
  const int wave  = tid >> 5;
  const int hi    = lane >> 4;
  const int l16   = lane & 15;
  const int h     = blockIdx.y;
  const int qtile = blockIdx.x * 128 + wave * 16;
  const int hb    = h * HD;

  // Fill-side addressing for this thread's two 16B chunks per K-step.
  const int kks = tid >> 7;               // which K=32 half of head dim
  const int kn  = (tid >> 6) & 1;         // which 16-key half
  const int fr  = tid & 63;
  const int fl  = fr >> 1;
  const int fh  = fr & 1;
  const bf16* ksrc = K + (kn * 16 + (fl & 15)) * DIM + hb + kks * 32
                       + (fl >> 4) * 8 + fh * 16;          // + kb*DIM per step
  const int vt_ = tid >> 6;               // V chunk: d-tile
  const bf16* vsrc = Vt + (hb + vt_ * 16 + (fl & 15)) * SEQ
                        + (fl >> 4) * 8 + fh * 16;          // + kb per step

  // Preload Q A-operands (head dim 64 = two K=32 steps); reused whole loop.
  v16bf aq[2];
#pragma unroll
  for (int ks = 0; ks < 2; ++ks) {
    const bf16* qrow = Q + (qtile + l16) * DIM + hb + ks * 32 + hi * 8;
    aq[ks] = combine16(qrow, qrow + 16);
  }

  int qmr[8];
#pragma unroll
  for (int i = 0; i < 8; ++i) qmr[i] = msk[qtile + i + hi * 8];

  float m_i[8], l_i[8];
  v8f acc[4];
#pragma unroll
  for (int i = 0; i < 8; ++i) { m_i[i] = -3.0e38f; l_i[i] = 0.0f; }
#pragma unroll
  for (int t = 0; t < 4; ++t) acc[t] = zero8();

  issue_async_b128(lds_off32(&skv[0][tid * 8]), gaddr64(ksrc));
  issue_async_b128(lds_off32(&skv[0][(256 + tid) * 8]), gaddr64(vsrc));

#pragma unroll 2
  for (int it = 0; it < SEQ / 32; ++it) {
    const int kb = it * 32;
    const bf16* cur = skv[it & 1];
    wait_async0();
    __syncthreads();
    if (kb + 32 < SEQ) {
      bf16* nxt = skv[(it + 1) & 1];
      issue_async_b128(lds_off32(nxt + tid * 8),
                       gaddr64(ksrc + (size_t)(kb + 32) * DIM));
      issue_async_b128(lds_off32(nxt + (256 + tid) * 8),
                       gaddr64(vsrc + kb + 32));
    }

    // Hoist all B operands (4x K, 4x V) out of LDS before the WMMA bursts.
    v16bf bk_[4], bv_[4];
#pragma unroll
    for (int j = 0; j < 4; ++j) {
      const bf16* p = cur + (j * 64 + lane * 2) * 8;
      bk_[j] = combine16(p, p + 8);
    }
#pragma unroll
    for (int t = 0; t < 4; ++t) {
      const bf16* p = cur + 2048 + (t * 64 + lane * 2) * 8;
      bv_[t] = combine16(p, p + 8);
    }

    // S = Q K^T for 32 keys -> two 16x16 f32 tiles
    v8f s0 = zero8(), s1 = zero8();
    s0 = wmma_bf16(aq[0], bk_[0], s0);
    s1 = wmma_bf16(aq[0], bk_[1], s1);
    s0 = wmma_bf16(aq[1], bk_[2], s0);
    s1 = wmma_bf16(aq[1], bk_[3], s1);

    const int km0 = msk[kb + l16];
    const int km1 = msk[kb + 16 + l16];

    float p0[8], p1[8], alpha[8];
#pragma unroll
    for (int i = 0; i < 8; ++i) {
      float f0 = s0[i] * 0.125f;               // 1/sqrt(64)
      float f1 = s1[i] * 0.125f;
      const bool qok = (qmr[i] != 0);
      if (!(qok && km0)) f0 = -3.0e38f;
      if (!(qok && km1)) f1 = -3.0e38f;
      float tv = fmaxf(f0, f1);
#pragma unroll
      for (int off = 8; off > 0; off >>= 1)
        tv = fmaxf(tv, __shfl_xor(tv, off, 16));
      const float mnew = fmaxf(m_i[i], tv);
      const float al = __expf(m_i[i] - mnew);
      const float e0 = __expf(f0 - mnew);
      const float e1 = __expf(f1 - mnew);
      float rs = e0 + e1;
#pragma unroll
      for (int off = 8; off > 0; off >>= 1) rs += __shfl_xor(rs, off, 16);
      l_i[i] = l_i[i] * al + rs;
      m_i[i] = mnew;
      alpha[i] = al;
      p0[i] = e0;
      p1[i] = e1;
    }
#pragma unroll
    for (int t = 0; t < 4; ++t)
#pragma unroll
      for (int i = 0; i < 8; ++i) acc[t][i] *= alpha[i];

    // Repack P from C/D layout to A layout via per-wave LDS slice.
    // Per-wave LDS is processed in order (DScnt in-order), so the
    // lane-crossing RAW needs only a compiler ordering barrier.
#pragma unroll
    for (int i = 0; i < 8; ++i) {
      plds[wave][(i + hi * 8) * 32 + l16]      = (bf16)p0[i];
      plds[wave][(i + hi * 8) * 32 + 16 + l16] = (bf16)p1[i];
    }
    asm volatile("" ::: "memory");
    v16bf pa = combine16(&plds[wave][l16 * 32 + hi * 8],
                         &plds[wave][l16 * 32 + 16 + hi * 8]);

    // acc += P @ V
#pragma unroll
    for (int t = 0; t < 4; ++t) acc[t] = wmma_bf16(pa, bv_[t], acc[t]);
  }

  float rl[8];
#pragma unroll
  for (int i = 0; i < 8; ++i) rl[i] = 1.0f / l_i[i];
#pragma unroll
  for (int t = 0; t < 4; ++t) {
    const int col = hb + t * 16 + l16;
#pragma unroll
    for (int i = 0; i < 8; ++i) {
      const int row = qtile + i + hi * 8;
      Aout[row * DIM + col] = (bf16)(acc[t][i] * rl[i]);
    }
  }
}

// ---------------------------------------------------------------------------
// Kernel 4: output projection + bias + residual -> f32.
// ---------------------------------------------------------------------------
__global__ __launch_bounds__(256)
void proj_gemm_kernel(const bf16* __restrict__ Aat, const bf16* __restrict__ Wo,
                      const float* __restrict__ bo,
                      const float* __restrict__ resid, float* __restrict__ out) {
  __shared__ bf16 sB[2 * 2048];
  const int lane = threadIdx.x & 31;
  const int wave = threadIdx.x >> 5;
  const int hi   = lane >> 4;
  const int l16  = lane & 15;

  const int mtile = blockIdx.y * 128 + wave * 16;
  const int nbase = blockIdx.x * 64;

  v8f acc[4];
#pragma unroll
  for (int t = 0; t < 4; ++t) acc[t] = zero8();

  gemm_loop(Aat + (mtile + l16) * DIM, Wo, nbase, sB, acc);

#pragma unroll
  for (int t = 0; t < 4; ++t) {
    const int col = nbase + t * 16 + l16;
    const float bb = bo[col];
#pragma unroll
    for (int i = 0; i < 8; ++i) {
      const int row = mtile + i + hi * 8;
      out[row * DIM + col] = acc[t][i] + bb + resid[row * DIM + col];
    }
  }
}

// ---------------------------------------------------------------------------
extern "C" void kernel_launch(void* const* d_in, const int* in_sizes, int n_in,
                              void* d_out, int out_size, void* d_ws,
                              size_t ws_size, hipStream_t stream) {
  const float* cs  = (const float*)d_in[0];
  const float* sn  = (const float*)d_in[1];
  const float* x   = (const float*)d_in[2];
  const int*   msk = (const int*)d_in[3];
  const float* Wq  = (const float*)d_in[4];
  const float* bq  = (const float*)d_in[5];
  const float* Wk  = (const float*)d_in[6];
  const float* bk  = (const float*)d_in[7];
  const float* Wv  = (const float*)d_in[8];
  const float* bv  = (const float*)d_in[9];
  const float* Wo  = (const float*)d_in[10];
  const float* bo  = (const float*)d_in[11];
  float* out = (float*)d_out;

  char* ws = (char*)d_ws;
  bf16* xr  = (bf16*)ws;  ws += (size_t)SEQ * DIM * 2;   // roped input, bf16
  bf16* wqb = (bf16*)ws;  ws += (size_t)DIM * DIM * 2;
  bf16* wkb = (bf16*)ws;  ws += (size_t)DIM * DIM * 2;
  bf16* wvb = (bf16*)ws;  ws += (size_t)DIM * DIM * 2;
  bf16* wob = (bf16*)ws;  ws += (size_t)DIM * DIM * 2;
  bf16* Qb  = (bf16*)ws;  ws += (size_t)SEQ * DIM * 2;
  bf16* Kb  = (bf16*)ws;  ws += (size_t)SEQ * DIM * 2;
  bf16* Vtb = (bf16*)ws;  ws += (size_t)SEQ * DIM * 2;   // transposed V
  bf16* Ab  = (bf16*)ws;  ws += (size_t)SEQ * DIM * 2;   // attention output

  rope_bf16_kernel<<<(SEQ * DIM) / 256, 256, 0, stream>>>(x, cs, sn, xr);
  cvt_bf16_kernel<<<(DIM * DIM) / 256, 256, 0, stream>>>(Wq, wqb);
  cvt_bf16_kernel<<<(DIM * DIM) / 256, 256, 0, stream>>>(Wk, wkb);
  cvt_bf16_kernel<<<(DIM * DIM) / 256, 256, 0, stream>>>(Wv, wvb);
  cvt_bf16_kernel<<<(DIM * DIM) / 256, 256, 0, stream>>>(Wo, wob);

  dim3 gq(DIM / 64, SEQ / 128, 3);
  qkv_gemm_kernel<<<gq, 256, 0, stream>>>(xr, wqb, wkb, wvb, bq, bk, bv, Qb,
                                          Kb, Vtb);

  dim3 ga(SEQ / 128, HEADS);
  flash_attn_kernel<<<ga, 256, 0, stream>>>(Qb, Kb, Vtb, msk, Ab);

  dim3 gp(DIM / 64, SEQ / 128);
  proj_gemm_kernel<<<gp, 256, 0, stream>>>(Ab, wob, bo, x, out);
}
